// MultiHeadAttention_34359738768
// MI455X (gfx1250) — compile-verified, matrix operand order checked by reading, not run on hardware
//
#include <hip/hip_runtime.h>
#include <stdint.h>

typedef __attribute__((ext_vector_type(16))) __bf16 v16bf;
typedef __attribute__((ext_vector_type(8)))  float  v8f;

#define B_  2
#define S_  2048
#define D_  1024
#define H_  16
#define HD_ 64

union Frag16 {
    v16bf v;
    uint4 q[2];
    uint32_t u[8];
};

static __device__ __forceinline__ uint16_t f2bf(float f) {
    uint32_t u = __float_as_uint(f);
    u += 0x7FFFu + ((u >> 16) & 1u);        // round-to-nearest-even
    return (uint16_t)(u >> 16);
}

// ---------------------------------------------------------------- convert
__global__ void f32_to_bf16_kernel(const float* __restrict__ in,
                                   uint16_t* __restrict__ out, int n) {
    for (int i = blockIdx.x * blockDim.x + threadIdx.x; i < n;
         i += gridDim.x * blockDim.x)
        out[i] = f2bf(in[i]);
}

// ---------------------------------------------------------------- GEMM y = A * W^T + bias
// A: [4096, 1024] bf16 row-major, Wt: [1024 (n), 1024 (k)] bf16 row-major.
// Block: 256 threads = 8 waves arranged 4(M) x 2(N); block tile 64M x 128N.
// Wave tile: 16M x 64N = 4 f32 accumulators; K loop in steps of 32.
enum { MODE_HEADS = 0, MODE_VT = 1, MODE_F32 = 2 };

template <int MODE>
__global__ __launch_bounds__(256) void gemm_nt_wmma(
    const uint16_t* __restrict__ A, const uint16_t* __restrict__ Wt,
    const float* __restrict__ bias, void* __restrict__ outp, float scale) {
    const int lane = threadIdx.x & 31;
    const int wave = threadIdx.x >> 5;
    const int wm = wave & 3, wn = wave >> 2;
    const int m0 = blockIdx.x * 64 + wm * 16;
    const int n0 = blockIdx.y * 128 + wn * 64;
    const int row = lane & 15;
    const int hf  = lane >> 4;           // which 16-lane half

    const uint16_t* arow = A + (size_t)(m0 + row) * D_;
    const uint16_t* brow[4];
#pragma unroll
    for (int t = 0; t < 4; ++t)
        brow[t] = Wt + (size_t)(n0 + t * 16 + row) * D_;

    v8f acc[4] = {};

    for (int k0 = 0; k0 < D_; k0 += 32) {
        if (k0 + 32 < D_) {                       // hide next K-slab latency
            __builtin_prefetch(arow + k0 + 32, 0, 3);
            __builtin_prefetch(brow[0] + k0 + 32, 0, 3);
        }
        Frag16 a;
        a.q[0] = *(const uint4*)(arow + k0 + hf * 8);
        a.q[1] = *(const uint4*)(arow + k0 + 16 + hf * 8);
#pragma unroll
        for (int t = 0; t < 4; ++t) {
            Frag16 b;
            b.q[0] = *(const uint4*)(brow[t] + k0 + hf * 8);
            b.q[1] = *(const uint4*)(brow[t] + k0 + 16 + hf * 8);
            acc[t] = __builtin_amdgcn_wmma_f32_16x16x32_bf16(
                false, a.v, false, b.v, (short)0, acc[t], false, false);
        }
    }

#pragma unroll
    for (int t = 0; t < 4; ++t) {
        const int gn = n0 + t * 16 + row;
        const float bias_n = bias[gn];
#pragma unroll
        for (int r = 0; r < 8; ++r) {
            const int gm = m0 + hf * 8 + r;
            const float val = (acc[t][r] + bias_n) * scale;
            if (MODE == MODE_HEADS) {           // [B,H,S,HD] bf16
                const int b = gm >> 11, s = gm & (S_ - 1);
                const int h = gn >> 6, hd = gn & (HD_ - 1);
                ((uint16_t*)outp)[((size_t)(b * H_ + h) * S_ + s) * HD_ + hd] = f2bf(val);
            } else if (MODE == MODE_VT) {       // [B,H,HD,S] bf16 (V transposed)
                const int b = gm >> 11, s = gm & (S_ - 1);
                const int h = gn >> 6, hd = gn & (HD_ - 1);
                ((uint16_t*)outp)[((size_t)(b * H_ + h) * HD_ + hd) * S_ + s] = f2bf(val);
            } else {                            // plain fp32 row-major
                ((float*)outp)[(size_t)gm * D_ + gn] = val;
            }
        }
    }
}

// ---------------------------------------------------------------- flash attention
// Grid: (S/64, B*H); 128 threads = 4 waves; each wave owns one 16-row Q tile.
// Key chunk = 64: 8 score WMMAs + one softmax/rescale pass + 8 P*V WMMAs +
// 2 P*1 WMMAs (row sums via matrix pipe; no sum shuffles).
__global__ __launch_bounds__(128) void flash_attn_wmma(
    const uint16_t* __restrict__ qh, const uint16_t* __restrict__ kh,
    const uint16_t* __restrict__ vt, uint16_t* __restrict__ oh) {
    __shared__ __align__(16) uint16_t plds[4][16 * 64];   // 2KB per wave

    const int lane = threadIdx.x & 31;
    const int wave = threadIdx.x >> 5;
    const int row  = lane & 15;
    const int hf   = lane >> 4;
    const int bh   = blockIdx.y;
    const int q0   = blockIdx.x * 64 + wave * 16;

    const uint16_t* qb  = qh + (size_t)bh * S_ * HD_;
    const uint16_t* kbp = kh + (size_t)bh * S_ * HD_;
    const uint16_t* vb  = vt + (size_t)bh * HD_ * S_;

    // Q fragments over HD=64: two K-steps of 32 (already scaled by 1/sqrt(HD))
    const uint16_t* qrow = qb + (size_t)(q0 + row) * HD_;
    Frag16 aq0, aq1;
    aq0.q[0] = *(const uint4*)(qrow + hf * 8);
    aq0.q[1] = *(const uint4*)(qrow + 16 + hf * 8);
    aq1.q[0] = *(const uint4*)(qrow + 32 + hf * 8);
    aq1.q[1] = *(const uint4*)(qrow + 48 + hf * 8);

    // Constant all-ones B fragment (bf16 1.0 packed twice): row sums via WMMA
    Frag16 ones;
#pragma unroll
    for (int i = 0; i < 8; ++i) ones.u[i] = 0x3F803F80u;

    v8f oacc[4] = {};
    v8f lacc = {};                      // running row sums (replicated over lanes)
    float mrow[8];
#pragma unroll
    for (int r = 0; r < 8; ++r) mrow[r] = -3.0e38f;

    uint16_t* myl = plds[wave];

    for (int kb = 0; kb < S_; kb += 64) {
        // ---- scores: 4 tiles of 16 key columns, K dim = HD = 2 x 32
        v8f sc[4] = {};
#pragma unroll
        for (int tn = 0; tn < 4; ++tn) {
            const uint16_t* kr = kbp + (size_t)(kb + tn * 16 + row) * HD_;
            Frag16 b;
            b.q[0] = *(const uint4*)(kr + hf * 8);
            b.q[1] = *(const uint4*)(kr + 16 + hf * 8);
            sc[tn] = __builtin_amdgcn_wmma_f32_16x16x32_bf16(
                false, aq0.v, false, b.v, (short)0, sc[tn], false, false);
            b.q[0] = *(const uint4*)(kr + 32 + hf * 8);
            b.q[1] = *(const uint4*)(kr + 48 + hf * 8);
            sc[tn] = __builtin_amdgcn_wmma_f32_16x16x32_bf16(
                false, aq1.v, false, b.v, (short)0, sc[tn], false, false);
        }

        // ---- online softmax over 64 new key columns.
        // C-layout: n = lane&15 per half, rows m = hf*8 + r.
#pragma unroll
        for (int r = 0; r < 8; ++r) {
            float lm = fmaxf(fmaxf(sc[0][r], sc[1][r]), fmaxf(sc[2][r], sc[3][r]));
            lm = fmaxf(lm, __shfl_xor(lm, 1));
            lm = fmaxf(lm, __shfl_xor(lm, 2));
            lm = fmaxf(lm, __shfl_xor(lm, 4));
            lm = fmaxf(lm, __shfl_xor(lm, 8));
            const float mnew  = fmaxf(mrow[r], lm);
            const float alpha = __expf(mrow[r] - mnew);
            mrow[r] = mnew;
            lacc[r] *= alpha;
#pragma unroll
            for (int t = 0; t < 4; ++t) oacc[t][r] *= alpha;
            const int m = hf * 8 + r;
#pragma unroll
            for (int tn = 0; tn < 4; ++tn)
                myl[m * 64 + tn * 16 + row] = f2bf(__expf(sc[tn][r] - mnew));
        }
        __syncthreads();

        // ---- reload P as two A-fragments (16x64 -> 2 x K32)
        const uint16_t* pl = myl + row * 64;
        Frag16 pf0, pf1;
        pf0.q[0] = *(const uint4*)(pl + hf * 8);
        pf0.q[1] = *(const uint4*)(pl + 16 + hf * 8);
        pf1.q[0] = *(const uint4*)(pl + 32 + hf * 8);
        pf1.q[1] = *(const uint4*)(pl + 48 + hf * 8);
        __syncthreads();

        // ---- row sums: l += P * 1 (matrix pipe, no shuffles)
        lacc = __builtin_amdgcn_wmma_f32_16x16x32_bf16(
            false, pf0.v, false, ones.v, (short)0, lacc, false, false);
        lacc = __builtin_amdgcn_wmma_f32_16x16x32_bf16(
            false, pf1.v, false, ones.v, (short)0, lacc, false, false);

        // ---- O += P * V ; B[k][n] = Vt[n][kb+k], contiguous per lane
#pragma unroll
        for (int t = 0; t < 4; ++t) {
            const uint16_t* vr = vb + (size_t)(t * 16 + row) * S_ + kb;
            Frag16 b;
            b.q[0] = *(const uint4*)(vr + hf * 8);
            b.q[1] = *(const uint4*)(vr + 16 + hf * 8);
            oacc[t] = __builtin_amdgcn_wmma_f32_16x16x32_bf16(
                false, pf0.v, false, b.v, (short)0, oacc[t], false, false);
            b.q[0] = *(const uint4*)(vr + 32 + hf * 8);
            b.q[1] = *(const uint4*)(vr + 48 + hf * 8);
            oacc[t] = __builtin_amdgcn_wmma_f32_16x16x32_bf16(
                false, pf1.v, false, b.v, (short)0, oacc[t], false, false);
        }
    }

    // Epilogue: O / l, store as bf16 into [B,S,D] layout (heads merged back)
    const int bb = bh / H_, h = bh % H_;
#pragma unroll
    for (int r = 0; r < 8; ++r) {
        const float inv = 1.0f / lacc[r];
        const int s = q0 + hf * 8 + r;
#pragma unroll
        for (int t = 0; t < 4; ++t) {
            const int col = h * HD_ + t * 16 + row;
            oh[(size_t)(bb * S_ + s) * D_ + col] = f2bf(oacc[t][r] * inv);
        }
    }
}

// ---------------------------------------------------------------- launch
extern "C" void kernel_launch(void* const* d_in, const int* in_sizes, int n_in,
                              void* d_out, int out_size, void* d_ws, size_t ws_size,
                              hipStream_t stream) {
    const float* x  = (const float*)d_in[0];
    const float* Wq = (const float*)d_in[1];
    const float* bq = (const float*)d_in[2];
    const float* Wk = (const float*)d_in[3];
    const float* bk = (const float*)d_in[4];
    const float* Wv = (const float*)d_in[5];
    const float* bv = (const float*)d_in[6];
    const float* Wo = (const float*)d_in[7];
    const float* bo = (const float*)d_in[8];
    float* out = (float*)d_out;

    uint8_t* ws = (uint8_t*)d_ws;
    const size_t MB = 1024 * 1024;
    uint16_t* xb  = (uint16_t*)(ws + 0 * MB);   // 8 MB  x as bf16
    uint16_t* wqb = (uint16_t*)(ws + 8 * MB);   // 2 MB
    uint16_t* wkb = (uint16_t*)(ws + 10 * MB);  // 2 MB
    uint16_t* wvb = (uint16_t*)(ws + 12 * MB);  // 2 MB
    uint16_t* wob = (uint16_t*)(ws + 14 * MB);  // 2 MB
    uint16_t* qhd = (uint16_t*)(ws + 16 * MB);  // 8 MB  Q [B,H,S,HD] (pre-scaled)
    uint16_t* khd = (uint16_t*)(ws + 24 * MB);  // 8 MB  K [B,H,S,HD]
    uint16_t* vtr = (uint16_t*)(ws + 32 * MB);  // 8 MB  V^T [B,H,HD,S]
    uint16_t* ohd = (uint16_t*)(ws + 40 * MB);  // 8 MB  attn out [B,S,D] bf16

    const int XN = B_ * S_ * D_;
    const int WN = D_ * D_;

    f32_to_bf16_kernel<<<1024, 256, 0, stream>>>(x,  xb,  XN);
    f32_to_bf16_kernel<<<512,  256, 0, stream>>>(Wq, wqb, WN);
    f32_to_bf16_kernel<<<512,  256, 0, stream>>>(Wk, wkb, WN);
    f32_to_bf16_kernel<<<512,  256, 0, stream>>>(Wv, wvb, WN);
    f32_to_bf16_kernel<<<512,  256, 0, stream>>>(Wo, wob, WN);

    dim3 ggrid(B_ * S_ / 64, D_ / 128);
    gemm_nt_wmma<MODE_HEADS><<<ggrid, 256, 0, stream>>>(xb, wqb, bq, qhd, 0.125f); // 1/sqrt(64)
    gemm_nt_wmma<MODE_HEADS><<<ggrid, 256, 0, stream>>>(xb, wkb, bk, khd, 1.0f);
    gemm_nt_wmma<MODE_VT>   <<<ggrid, 256, 0, stream>>>(xb, wvb, bv, vtr, 1.0f);

    flash_attn_wmma<<<dim3(S_ / 64, B_ * H_), 128, 0, stream>>>(qhd, khd, vtr, ohd);

    gemm_nt_wmma<MODE_F32>  <<<ggrid, 256, 0, stream>>>(ohd, wob, bo, out, 1.0f);
}
